// SelfAttnBlock_4449586119257
// MI455X (gfx1250) — matrix operand order repaired and checked by reading, not compiled, not run on hardware
//
#include <hip/hip_runtime.h>

typedef __attribute__((ext_vector_type(16))) __bf16          v16bf;
typedef __attribute__((ext_vector_type(8)))  float           v8f;
typedef __attribute__((ext_vector_type(8)))  unsigned short  us8;
typedef __attribute__((ext_vector_type(16))) unsigned short  us16;
typedef __attribute__((ext_vector_type(4)))  unsigned short  us4;
typedef __attribute__((ext_vector_type(4)))  int             v4i;

#define DEV static __device__ __forceinline__

// ---------------------------------------------------------------- shapes -----
constexpr int B_ = 2, L_ = 2048, C_ = 2048, H_ = 16, D_ = 128;
constexpr int COND_ = 2048, HID_ = 8192;
constexpr int C3_ = 3 * C_, C6_ = 6 * C_, M_ = B_ * L_;
constexpr int LOG2L = 11;  // L_ = 2048

// ------------------------------------------------- gfx1250 async LDS staging --
#if __has_builtin(__builtin_amdgcn_global_load_async_to_lds_b128) && \
    __has_builtin(__builtin_amdgcn_s_wait_asynccnt)
#define HAVE_ASYNC 1
#else
#define HAVE_ASYNC 0
#endif

#if HAVE_ASYNC
typedef __attribute__((address_space(1))) v4i gv4i;  // global
typedef __attribute__((address_space(3))) v4i lv4i;  // LDS
DEV void async_ld128(const unsigned short* g, unsigned short* l) {
  __builtin_amdgcn_global_load_async_to_lds_b128((gv4i*)g, (lv4i*)l, 0, 0);
}
#define ASYNC_WAIT(n) __builtin_amdgcn_s_wait_asynccnt(n)
#else
DEV void async_ld128(const unsigned short* g, unsigned short* l) {
  *(us8*)l = *(const us8*)g;
}
#define ASYNC_WAIT(n) ((void)0)
#endif

DEV unsigned short f2bf(float f) {
  unsigned u = __builtin_bit_cast(unsigned, f);
  u += 0x7FFFu + ((u >> 16) & 1u);  // round-to-nearest-even
  return (unsigned short)(u >> 16);
}
DEV v8f zero8() {
  v8f z;
#pragma unroll
  for (int i = 0; i < 8; i++) z[i] = 0.f;
  return z;
}

// A-matrix 16x32 bf16 fragment (ISA 7.12.2): lane l holds row (l&15);
// halfs 0..7 -> K = (l>=16?8:0)+0..7 ; halfs 8..15 -> K = 16+(l>=16?8:0)+0..7
DEV v16bf frag_a(const unsigned short* base, int stride) {
  int l = threadIdx.x & 31;
  const unsigned short* p = base + (l & 15) * stride + ((l >> 4) << 3);
  us8 lo = *(const us8*)(p);
  us8 hh = *(const us8*)(p + 16);
  us16 f;
#pragma unroll
  for (int i = 0; i < 8; i++) { f[i] = lo[i]; f[8 + i] = hh[i]; }
  return __builtin_bit_cast(v16bf, f);
}

// B-matrix 32x16 bf16 fragment: lane l holds column (l&15);
// halfs 0..7 -> K = (l>=16?8:0)+0..7 ; halfs 8..15 -> K = 16+(l>=16?8:0)+0..7. Source tile is [N][K] row-major.
DEV v16bf frag_b(const unsigned short* base, int stride) {
  int l = threadIdx.x & 31;
  const unsigned short* p = base + (l & 15) * stride + ((l >> 4) << 3);
  us8 lo = *(const us8*)(p);
  us8 hh = *(const us8*)(p + 16);
  us16 f;
#pragma unroll
  for (int i = 0; i < 8; i++) { f[i] = lo[i]; f[8 + i] = hh[i]; }
  return __builtin_bit_cast(v16bf, f);
}

DEV v8f wmma_bf(v16bf a, v16bf b, v8f c) {
  return __builtin_amdgcn_wmma_f32_16x16x32_bf16(false, a, false, b, (short)0, c,
                                                 false, false);
}

// ---------------------------------------------------------------- fp32 -> bf16
__global__ __launch_bounds__(256) void k_cvt(const float* __restrict__ in,
                                             unsigned short* __restrict__ out,
                                             int n) {
  int i = blockIdx.x * 256 + threadIdx.x;
  if (i < n) out[i] = f2bf(in[i]);
}

// ------------------------------------------------- ada = silu(cond)@Wada^T + b
__global__ __launch_bounds__(256) void k_ada(const float* __restrict__ cond,
                                             const float* __restrict__ Wada,
                                             const float* __restrict__ bada,
                                             float* __restrict__ ada) {
  __shared__ float sc[COND_];
  int b = blockIdx.y, tid = threadIdx.x;
  for (int k = tid; k < COND_; k += 256) {
    float v = cond[(size_t)b * COND_ + k];
    sc[k] = v / (1.f + expf(-v));
  }
  __syncthreads();
  int o = blockIdx.x * 256 + tid;
  const float* wr = Wada + (size_t)o * COND_;
  float acc = bada[o];
  for (int k = 0; k < COND_; k += 4) {
    float4 w = *(const float4*)(wr + k);
    acc += w.x * sc[k] + w.y * sc[k + 1] + w.z * sc[k + 2] + w.w * sc[k + 3];
  }
  ada[(size_t)b * C6_ + o] = acc;
}

// -------------------------------- out_bf16 = LN(x) * (1+scale) + shift  (AdaLN)
__global__ __launch_bounds__(256) void k_ln_mod(const float* __restrict__ x,
                                                const float* __restrict__ ada,
                                                int scale_i, int shift_i,
                                                unsigned short* __restrict__ out) {
  int m = blockIdx.x, tid = threadIdx.x;
  const float* xr = x + (size_t)m * C_;
  float s = 0.f, s2 = 0.f;
  for (int c = tid; c < C_; c += 256) {
    float v = xr[c];
    s += v;
    s2 += v * v;
  }
#pragma unroll
  for (int o = 16; o; o >>= 1) {
    s += __shfl_xor(s, o);
    s2 += __shfl_xor(s2, o);
  }
  __shared__ float red[16];
  __shared__ float mv[2];
  int wid = tid >> 5, lane = tid & 31;
  if (lane == 0) { red[wid] = s; red[8 + wid] = s2; }
  __syncthreads();
  if (tid == 0) {
    float a = 0.f, bb = 0.f;
    for (int i = 0; i < 8; i++) { a += red[i]; bb += red[8 + i]; }
    float mean = a * (1.f / C_);
    mv[0] = mean;
    mv[1] = rsqrtf(bb * (1.f / C_) - mean * mean + 1e-6f);
  }
  __syncthreads();
  float mean = mv[0], rstd = mv[1];
  int b = m >> LOG2L;
  const float* sc = ada + (size_t)b * C6_ + (size_t)scale_i * C_;
  const float* sh = ada + (size_t)b * C6_ + (size_t)shift_i * C_;
  unsigned short* orow = out + (size_t)m * C_;
  for (int c = tid; c < C_; c += 256)
    orow[c] = f2bf((xr[c] - mean) * rstd * (1.f + sc[c]) + sh[c]);
}

// ------------------------------------------------------------ generic WMMA GEMM
// out[M,N] = A[M,K](bf16) @ W[N,K]^T(bf16); double-buffered async LDS staging.
// EPI 0: outF = acc                       (fp32, qkv)
// EPI 1: outF = resid + (acc+bias)*gamma  (fp32)
// EPI 2: outB = bf16(gelu_tanh(acc+bias))
template <int N, int K, int EPI>
__global__ __launch_bounds__(256) void k_gemm(
    const unsigned short* __restrict__ A, const unsigned short* __restrict__ W,
    const float* __restrict__ bias, const float* __restrict__ resid,
    const float* __restrict__ gamma, float* __restrict__ outF,
    unsigned short* __restrict__ outB) {
  constexpr int BK = 64, AST = BK + 8;  // pad 8 halfs
  extern __shared__ unsigned short smem[];
  auto sAp = [&](int buf) { return smem + buf * (128 * AST); };
  auto sWp = [&](int buf) { return smem + (2 + buf) * (128 * AST); };
  int tid = threadIdx.x, lane = tid & 31, wid = tid >> 5;
  int wm = wid & 3, wn = wid >> 2;
  int bm = blockIdx.x, bn = blockIdx.y;

  v8f acc[2][4];
#pragma unroll
  for (int i = 0; i < 2; i++)
#pragma unroll
    for (int j = 0; j < 4; j++) acc[i][j] = zero8();

  int r = tid >> 1, csb = (tid & 1) * 32;
  const unsigned short* gA = A + (size_t)(bm * 128 + r) * K + csb;
  const unsigned short* gW = W + (size_t)(bn * 128 + r) * K + csb;

  auto stage = [&](int s, int buf) {
    const unsigned short* ga = gA + (size_t)s * BK;
    const unsigned short* gw = gW + (size_t)s * BK;
    unsigned short* la = sAp(buf) + r * AST + csb;
    unsigned short* lw = sWp(buf) + r * AST + csb;
#pragma unroll
    for (int i = 0; i < 4; i++) {
      async_ld128(ga + i * 8, la + i * 8);
      async_ld128(gw + i * 8, lw + i * 8);
    }
  };

  constexpr int NS = K / BK;
  stage(0, 0);
  for (int s = 0; s < NS; ++s) {
    int buf = s & 1;
    if (s + 1 < NS) {
      stage(s + 1, buf ^ 1);
      ASYNC_WAIT(8);  // stage s complete; stage s+1 still in flight
    } else {
      ASYNC_WAIT(0);
    }
    __syncthreads();
    unsigned short* sAb = sAp(buf);
    unsigned short* sWb = sWp(buf);
#pragma unroll
    for (int kk = 0; kk < 2; ++kk) {
      v16bf af[2], bfr[4];
#pragma unroll
      for (int i = 0; i < 2; i++)
        af[i] = frag_a(sAb + (wm * 32 + i * 16) * AST + kk * 32, AST);
#pragma unroll
      for (int j = 0; j < 4; j++)
        bfr[j] = frag_b(sWb + (wn * 64 + j * 16) * AST + kk * 32, AST);
#pragma unroll
      for (int i = 0; i < 2; i++)
#pragma unroll
        for (int j = 0; j < 4; j++) acc[i][j] = wmma_bf(af[i], bfr[j], acc[i][j]);
    }
    __syncthreads();  // reads done before buf is overwritten two stages later
  }

  int nl = lane & 15, hi = lane >> 4;
#pragma unroll
  for (int i = 0; i < 2; i++)
#pragma unroll
    for (int j = 0; j < 4; j++)
#pragma unroll
      for (int rr = 0; rr < 8; rr++) {
        int m = bm * 128 + wm * 32 + i * 16 + hi * 8 + rr;
        int n = bn * 128 + wn * 64 + j * 16 + nl;
        float v = acc[i][j][rr];
        if constexpr (EPI == 0) {
          outF[(size_t)m * N + n] = v;
        } else if constexpr (EPI == 1) {
          v += bias[n];
          float g = gamma[(size_t)(m >> LOG2L) * C6_ + n];
          outF[(size_t)m * N + n] = resid[(size_t)m * N + n] + v * g;
        } else {
          v += bias[n];
          float t = 0.7978845608f * (v + 0.044715f * v * v * v);
          outB[(size_t)m * N + n] = f2bf(0.5f * v * (1.f + tanhf(t)));
        }
      }
}

// --------------------- qkv bias + cos-attn normalize; v stored transposed [BHDL]
__global__ __launch_bounds__(256) void k_qkvnorm(
    const float* __restrict__ qkv, const float* __restrict__ qb,
    const float* __restrict__ vb, const float* __restrict__ smlog,
    unsigned short* __restrict__ qn, unsigned short* __restrict__ kn,
    unsigned short* __restrict__ vt) {
  int wid = threadIdx.x >> 5, lane = threadIdx.x & 31;
  int idx = blockIdx.x * 8 + wid;  // (b, l, h) over B*L*H
  int h = idx & (H_ - 1);
  int l = (idx >> 4) & (L_ - 1);
  int b = idx >> (4 + LOG2L);
  size_t rb = ((size_t)(b * L_ + l) * 3) * C_ + h * D_ + lane * 4;
  float4 q = *(const float4*)(qkv + rb);
  float4 k4 = *(const float4*)(qkv + rb + C_);
  float4 v4 = *(const float4*)(qkv + rb + 2 * (size_t)C_);
  int cb = h * D_ + lane * 4;
  q.x += qb[cb]; q.y += qb[cb + 1]; q.z += qb[cb + 2]; q.w += qb[cb + 3];
  v4.x += vb[cb]; v4.y += vb[cb + 1]; v4.z += vb[cb + 2]; v4.w += vb[cb + 3];
  float qs = q.x * q.x + q.y * q.y + q.z * q.z + q.w * q.w;
  float ks = k4.x * k4.x + k4.y * k4.y + k4.z * k4.z + k4.w * k4.w;
#pragma unroll
  for (int o = 16; o; o >>= 1) {
    qs += __shfl_xor(qs, o);
    ks += __shfl_xor(ks, o);
  }
  float sm = expf(fminf(smlog[h], 4.6051702f));  // min(log-scale, log 100)
  float qd = sm / fmaxf(sqrtf(qs), 1e-12f);
  float kd = 1.f / fmaxf(sqrtf(ks), 1e-12f);
  size_t ob = ((size_t)(b * H_ + h) * L_ + l) * D_ + lane * 4;
  us4 qo, ko;
  qo[0] = f2bf(q.x * qd); qo[1] = f2bf(q.y * qd);
  qo[2] = f2bf(q.z * qd); qo[3] = f2bf(q.w * qd);
  ko[0] = f2bf(k4.x * kd); ko[1] = f2bf(k4.y * kd);
  ko[2] = f2bf(k4.z * kd); ko[3] = f2bf(k4.w * kd);
  *(us4*)(qn + ob) = qo;
  *(us4*)(kn + ob) = ko;
  size_t vtb = ((size_t)(b * H_ + h) * D_ + lane * 4) * L_ + l;
  vt[vtb] = f2bf(v4.x);
  vt[vtb + L_] = f2bf(v4.y);
  vt[vtb + 2 * (size_t)L_] = f2bf(v4.z);
  vt[vtb + 3 * (size_t)L_] = f2bf(v4.w);
}

// --------------------------------------------------- flash-style cos attention
// grid: (L/128, B*H). Scores bounded by sm -> exact softmax via constant shift.
// K/V tiles double-buffered via async LDS loads.
__global__ __launch_bounds__(256) void k_attn(
    const unsigned short* __restrict__ qn, const unsigned short* __restrict__ kn,
    const unsigned short* __restrict__ vt, const float* __restrict__ bias,
    const float* __restrict__ smlog, unsigned short* __restrict__ o) {
  constexpr int DSTR = 136;  // 128 + 8 halfs pad
  extern __shared__ unsigned short smem[];
  unsigned short* sQ = smem;
  auto sKp = [&](int buf) { return smem + (1 + buf) * (128 * DSTR); };
  auto sVp = [&](int buf) { return smem + (3 + buf) * (128 * DSTR); };
  unsigned short* sP = smem + 5 * (128 * DSTR);  // 8 waves * 16 rows
  int tid = threadIdx.x, lane = tid & 31, wid = tid >> 5;
  int nl = lane & 15, hi = lane >> 4;
  int qt = blockIdx.x, bh = blockIdx.y;
  int h = bh & (H_ - 1), b = bh >> 4;
  float sm = expf(fminf(smlog[h], 4.6051702f));

  int sr = tid >> 1, scs = (tid & 1) * 64;
  {  // stage Q tile (persistent)
    const unsigned short* g = qn + ((size_t)bh * L_ + qt * 128 + sr) * D_ + scs;
    unsigned short* d = sQ + sr * DSTR + scs;
#pragma unroll
    for (int i = 0; i < 8; i++) async_ld128(g + i * 8, d + i * 8);
  }
  auto stageKV = [&](int kt, int buf) {
    const unsigned short* gk = kn + ((size_t)bh * L_ + kt * 128 + sr) * D_ + scs;
    const unsigned short* gv = vt + ((size_t)bh * D_ + sr) * L_ + kt * 128 + scs;
    unsigned short* dk = sKp(buf) + sr * DSTR + scs;
    unsigned short* dv = sVp(buf) + sr * DSTR + scs;
#pragma unroll
    for (int i = 0; i < 8; i++) {
      async_ld128(gk + i * 8, dk + i * 8);
      async_ld128(gv + i * 8, dv + i * 8);
    }
  };
  stageKV(0, 0);

  v8f O[8];
#pragma unroll
  for (int j = 0; j < 8; j++) O[j] = zero8();
  float rowsum[8];
#pragma unroll
  for (int r = 0; r < 8; r++) rowsum[r] = 0.f;
  unsigned short* sPw = sP + wid * 16 * DSTR;

  constexpr int NKT = L_ / 128;
  for (int kt = 0; kt < NKT; ++kt) {
    int buf = kt & 1;
    if (kt + 1 < NKT) {
      stageKV(kt + 1, buf ^ 1);
      ASYNC_WAIT(16);  // Q + KV(kt) resident; KV(kt+1) in flight
    } else {
      ASYNC_WAIT(0);
    }
    __syncthreads();
    unsigned short* sKb = sKp(buf);
    unsigned short* sVb = sVp(buf);

    v8f S[8];
#pragma unroll
    for (int j = 0; j < 8; j++) S[j] = zero8();
#pragma unroll
    for (int ks = 0; ks < 4; ks++) {
      v16bf a = frag_a(sQ + (wid * 16) * DSTR + ks * 32, DSTR);
#pragma unroll
      for (int j = 0; j < 8; j++) {
        v16bf bb = frag_b(sKb + (j * 16) * DSTR + ks * 32, DSTR);
        S[j] = wmma_bf(a, bb, S[j]);
      }
    }
    // p = exp(score + bias - sm); write P tile (bf16) for A-fragments
#pragma unroll
    for (int j = 0; j < 8; j++)
#pragma unroll
      for (int r = 0; r < 8; r++) {
        int qg = qt * 128 + wid * 16 + hi * 8 + r;
        int kg = kt * 128 + j * 16 + nl;
        float p = expf(S[j][r] + bias[(size_t)qg * L_ + kg] - sm);
        S[j][r] = p;
        sPw[(hi * 8 + r) * DSTR + j * 16 + nl] = f2bf(p);
      }
#pragma unroll
    for (int r = 0; r < 8; r++) {
      float rs = 0.f;
#pragma unroll
      for (int j = 0; j < 8; j++) rs += S[j][r];
#pragma unroll
      for (int o2 = 8; o2; o2 >>= 1) rs += __shfl_xor(rs, o2);
      rowsum[r] += rs;
    }
    __syncthreads();  // sP visible wave-wide
#pragma unroll
    for (int ks = 0; ks < 4; ks++) {
      v16bf a = frag_a(sPw + ks * 32, DSTR);
#pragma unroll
      for (int j = 0; j < 8; j++) {
        v16bf bb = frag_b(sVb + (j * 16) * DSTR + ks * 32, DSTR);
        O[j] = wmma_bf(a, bb, O[j]);
      }
    }
    __syncthreads();  // done reading buf before it is restaged
  }

  float inv[8];
#pragma unroll
  for (int r = 0; r < 8; r++) inv[r] = 1.f / rowsum[r];
#pragma unroll
  for (int j = 0; j < 8; j++)
#pragma unroll
    for (int r = 0; r < 8; r++) {
      int qg = qt * 128 + wid * 16 + hi * 8 + r;
      int d = j * 16 + nl;
      o[((size_t)(b * L_) + qg) * C_ + h * D_ + d] = f2bf(O[j][r] * inv[r]);
    }
}

// ================================================================= launcher ==
extern "C" void kernel_launch(void* const* d_in, const int* in_sizes, int n_in,
                              void* d_out, int out_size, void* d_ws,
                              size_t ws_size, hipStream_t stream) {
  (void)in_sizes; (void)n_in; (void)out_size; (void)ws_size;
  const float* x = (const float*)d_in[0];
  const float* cond = (const float*)d_in[1];
  const float* abias = (const float*)d_in[2];
  const float* Wqkv = (const float*)d_in[3];
  const float* qb = (const float*)d_in[4];
  const float* vb = (const float*)d_in[5];
  const float* smlog = (const float*)d_in[6];
  const float* Wproj = (const float*)d_in[7];
  const float* bproj = (const float*)d_in[8];
  const float* Wada = (const float*)d_in[9];
  const float* bada = (const float*)d_in[10];
  const float* Wfc1 = (const float*)d_in[11];
  const float* bfc1 = (const float*)d_in[12];
  const float* Wfc2 = (const float*)d_in[13];
  const float* bfc2 = (const float*)d_in[14];
  float* out = (float*)d_out;

  char* p = (char*)d_ws;
  auto alloc = [&](size_t bytes) -> char* {
    char* q = p;
    p += (bytes + 255) & ~(size_t)255;
    return q;
  };
  float* ada = (float*)alloc((size_t)B_ * C6_ * 4);
  unsigned short* wqkv_b = (unsigned short*)alloc((size_t)C3_ * C_ * 2);
  unsigned short* wproj_b = (unsigned short*)alloc((size_t)C_ * C_ * 2);
  unsigned short* wfc1_b = (unsigned short*)alloc((size_t)HID_ * C_ * 2);
  unsigned short* wfc2_b = (unsigned short*)alloc((size_t)C_ * HID_ * 2);
  unsigned short* xin_b = (unsigned short*)alloc((size_t)M_ * C_ * 2);
  char* big = alloc((size_t)M_ * C3_ * 4);  // qkv fp32; later reused for o and h
  float* qkv_f = (float*)big;
  unsigned short* o_b = (unsigned short*)big;
  unsigned short* h_b =
      (unsigned short*)(big + (((size_t)M_ * C_ * 2 + 255) & ~(size_t)255));
  unsigned short* qn_b = (unsigned short*)alloc((size_t)B_ * H_ * L_ * D_ * 2);
  unsigned short* kn_b = (unsigned short*)alloc((size_t)B_ * H_ * L_ * D_ * 2);
  unsigned short* vt_b = (unsigned short*)alloc((size_t)B_ * H_ * L_ * D_ * 2);
  float* x1 = (float*)alloc((size_t)M_ * C_ * 4);

  auto cvt = [&](const float* in, unsigned short* o2, size_t n) {
    k_cvt<<<dim3((unsigned)((n + 255) / 256)), dim3(256), 0, stream>>>(in, o2,
                                                                       (int)n);
  };
  cvt(Wqkv, wqkv_b, (size_t)C3_ * C_);
  cvt(Wproj, wproj_b, (size_t)C_ * C_);
  cvt(Wfc1, wfc1_b, (size_t)HID_ * C_);
  cvt(Wfc2, wfc2_b, (size_t)C_ * HID_);

  k_ada<<<dim3(C6_ / 256, B_), 256, 0, stream>>>(cond, Wada, bada, ada);
  // LN1 + (scale1,shift1) = ada slots 2,4
  k_ln_mod<<<M_, 256, 0, stream>>>(x, ada, 2, 4, xin_b);

  size_t gemm_smem = (size_t)4 * 128 * 72 * 2;  // 73,728 B
  // qkv = xin @ Wqkv^T (bias folded into qkvnorm)
  k_gemm<C3_, C_, 0><<<dim3(M_ / 128, C3_ / 128), 256, gemm_smem, stream>>>(
      xin_b, wqkv_b, nullptr, nullptr, nullptr, qkv_f, nullptr);
  k_qkvnorm<<<(B_ * L_ * H_) / 8, 256, 0, stream>>>(qkv_f, qb, vb, smlog, qn_b,
                                                    kn_b, vt_b);
  size_t attn_smem = (size_t)6 * 128 * 136 * 2;  // Q, 2xK, 2xV, P = 208,896 B
  k_attn<<<dim3(L_ / 128, B_ * H_), 256, attn_smem, stream>>>(
      qn_b, kn_b, vt_b, abias, smlog, o_b);
  // x1 = x + (o @ Wproj^T + bproj) * gamma1 (slot 0)
  k_gemm<C_, C_, 1><<<dim3(M_ / 128, C_ / 128), 256, gemm_smem, stream>>>(
      o_b, wproj_b, bproj, x, ada + 0 * C_, x1, nullptr);
  // LN2 + (scale2,shift2) = slots 3,5
  k_ln_mod<<<M_, 256, 0, stream>>>(x1, ada, 3, 5, xin_b);
  // h = gelu(xin2 @ Wfc1^T + bfc1) -> bf16
  k_gemm<HID_, C_, 2><<<dim3(M_ / 128, HID_ / 128), 256, gemm_smem, stream>>>(
      xin_b, wfc1_b, bfc1, nullptr, nullptr, nullptr, h_b);
  // out = x1 + (h @ Wfc2^T + bfc2) * gamma2 (slot 1)
  k_gemm<C_, HID_, 1><<<dim3(M_ / 128, C_ / 128), 256, gemm_smem, stream>>>(
      h_b, wfc2_b, bfc2, x1, ada + 1 * C_, out, nullptr);
}
